// MLLABlock_46583215292777
// MI455X (gfx1250) — matrix operand order repaired and checked by reading, not compiled, not run on hardware
//
#include <hip/hip_runtime.h>
#include <math.h>

// ---------------------------------------------------------------------------
// MLLA block for MI455X (gfx1250, wave32).
//  - All dense GEMMs on v_wmma_f32_16x16x32_bf16 (f32 accumulate).
//  - GEMM weight tiles staged global->LDS via TDM (tensor_load_to_lds),
//    B fragments fed from LDS, waits via s_wait_tensorcnt.
//  - Elementwise stages fused into producers/consumers.
//  - Attention reductions (kmean, kv) chunked two-stage for latency hiding,
//    deterministic (no float atomics).
// ---------------------------------------------------------------------------

typedef __attribute__((ext_vector_type(8)))  __bf16 v8bf;
typedef __attribute__((ext_vector_type(16))) __bf16 v16bf;
typedef __attribute__((ext_vector_type(8)))  float  v8f;
typedef __attribute__((ext_vector_type(4)))  unsigned int u32x4;
typedef __attribute__((ext_vector_type(8)))  int i32x8;
typedef __attribute__((ext_vector_type(4)))  int i32x4;

#define SHUF16(lo,hi) __builtin_shufflevector((lo),(hi),0,1,2,3,4,5,6,7,8,9,10,11,12,13,14,15)

__device__ __forceinline__ v8f wmma_bf16(v16bf a, v16bf b, v8f c){
  // 8 args: (neg_a, A, neg_b, B, c_mod, C, reuse_a, reuse_b)
  return __builtin_amdgcn_wmma_f32_16x16x32_bf16(false, a, false, b, (short)0, c, false, false);
}

// ---------------------------------------------------------------------------
// TDM: DMA a 64-row x 32-element bf16 tile (row stride K elements) from a
// (N x K) bf16 matrix in global memory into LDS (contiguous 64x64B rows).
// Descriptor per CDNA5 ISA 8.3/8.4 (2-D tensor, groups 2/3 zero).
// 6-arg toolchain variant: (u32x4, i32x8, i32x4, i32x4, i32x8, i32 cpol)
// ---------------------------------------------------------------------------
__device__ __forceinline__ void tdm_load_tile_b(unsigned int lds_off,
                                                const __bf16* gptr,
                                                int K, int N){
  unsigned long long ga = (unsigned long long)(uintptr_t)gptr;
  u32x4 g0;
  g0[0] = 1u;                                        // count=1, user descriptor
  g0[1] = lds_off;                                   // LDS byte address
  g0[2] = (unsigned int)ga;                          // global_addr[31:0]
  g0[3] = ((unsigned int)(ga >> 32) & 0x01FFFFFFu)   // global_addr[56:32]
          | 0x80000000u;                             // type=2 ("image")
  i32x8 g1;
  g1[0] = 0x00010000;                                // data_size=1 (2 bytes)
  g1[1] = (int)((unsigned int)K << 16);              // tensor_dim0[15:0] @ [31:16]
  g1[2] = (int)(((unsigned int)N & 0xFFFFu) << 16);  // dim0 hi=0 | tensor_dim1 lo
  g1[3] = (int)(32u << 16);                          // dim1 hi=0 | tile_dim0=32
  g1[4] = 64;                                        // tile_dim1=64 | tile_dim2=0
  g1[5] = K;                                         // tensor_dim0_stride[31:0]
  g1[6] = 0;                                         // stride0 hi | stride1 lo
  g1[7] = 0;                                         // stride1 hi
  i32x4 gz4;
  gz4[0] = 0; gz4[1] = 0; gz4[2] = 0; gz4[3] = 0;    // groups 2/3 unused (2-D)
  i32x8 gz8;
  gz8[0] = 0; gz8[1] = 0; gz8[2] = 0; gz8[3] = 0;
  gz8[4] = 0; gz8[5] = 0; gz8[6] = 0; gz8[7] = 0;
  __builtin_amdgcn_tensor_load_to_lds(g0, g1, gz4, gz4, gz8, 0);
}

// ---------------------------------------------------------------------------
// Weight convert+transpose: W (K x N f32, row major) -> WT (N x K bf16)
// ---------------------------------------------------------------------------
__global__ __launch_bounds__(256) void wt_kernel(const float* __restrict__ W,
                                                 __bf16* __restrict__ WT,
                                                 int K, int N){
  size_t i = (size_t)blockIdx.x * 256 + threadIdx.x;
  size_t total = (size_t)K * N;
  if (i < total){
    int n = (int)(i % N);
    int k = (int)(i / N);
    WT[(size_t)n * K + k] = (__bf16)W[i];
  }
}

// ---------------------------------------------------------------------------
// CPE (3x3 depthwise + residual) fused with LayerNorm.
// ---------------------------------------------------------------------------
__global__ __launch_bounds__(256) void cpe_ln_kernel(
    const float* __restrict__ in, int nchw,
    const float* __restrict__ dw, const float* __restrict__ db,
    const float* __restrict__ g, const float* __restrict__ be,
    float* __restrict__ outT, __bf16* __restrict__ outN)
{
  int m = blockIdx.x;           // global token
  int c = threadIdx.x;          // channel
  int b = m >> 12, p = m & 4095;
  int y = p >> 6, x = p & 63;
  float acc = db[c];
  for (int dy = -1; dy <= 1; dy++){
    int yy = y + dy; if (yy < 0 || yy >= 64) continue;
    for (int dx = -1; dx <= 1; dx++){
      int xx = x + dx; if (xx < 0 || xx >= 64) continue;
      float v = nchw ? in[(((size_t)b * 256 + c) << 12) + yy * 64 + xx]
                     : in[((size_t)(b << 12) + yy * 64 + xx) * 256 + c];
      acc += v * dw[c * 9 + (dy + 1) * 3 + (dx + 1)];
    }
  }
  float center = nchw ? in[(((size_t)b * 256 + c) << 12) + p]
                      : in[(size_t)m * 256 + c];
  float t = center + acc;
  outT[(size_t)m * 256 + c] = t;
  // LayerNorm over C=256
  __shared__ float sh[256];
  sh[c] = t; __syncthreads();
  for (int s = 128; s > 0; s >>= 1){ if (c < s) sh[c] += sh[c + s]; __syncthreads(); }
  float mean = sh[0] * (1.f / 256.f); __syncthreads();
  float d = t - mean;
  sh[c] = d * d; __syncthreads();
  for (int s = 128; s > 0; s >>= 1){ if (c < s) sh[c] += sh[c + s]; __syncthreads(); }
  float var = sh[0] * (1.f / 256.f); __syncthreads();
  outN[(size_t)m * 256 + c] = (__bf16)(d * rsqrtf(var + 1e-5f) * g[c] + be[c]);
}

// ---------------------------------------------------------------------------
// Depthwise conv (no residual) + SiLU: ti (token-major f32) -> v f32 + vb bf16
// ---------------------------------------------------------------------------
__global__ __launch_bounds__(256) void dwc_silu_kernel(
    const float* __restrict__ in, const float* __restrict__ dw,
    const float* __restrict__ db, float* __restrict__ outF,
    __bf16* __restrict__ outB)
{
  int m = blockIdx.x, c = threadIdx.x;
  int b = m >> 12, p = m & 4095;
  int y = p >> 6, x = p & 63;
  float acc = db[c];
  for (int dy = -1; dy <= 1; dy++){
    int yy = y + dy; if (yy < 0 || yy >= 64) continue;
    for (int dx = -1; dx <= 1; dx++){
      int xx = x + dx; if (xx < 0 || xx >= 64) continue;
      acc += in[((size_t)(b << 12) + yy * 64 + xx) * 256 + c] *
             dw[c * 9 + (dy + 1) * 3 + (dx + 1)];
    }
  }
  float s = acc / (1.f + expf(-acc));     // SiLU
  outF[(size_t)m * 256 + c] = s;
  outB[(size_t)m * 256 + c] = (__bf16)s;
}

// ---------------------------------------------------------------------------
// kmean two-stage: partial sums over 128-token chunks, then reduce.
// kpart[b][ch][c], 32 chunks per batch. Deterministic (no atomics).
// ---------------------------------------------------------------------------
__global__ __launch_bounds__(256) void kmean_part_kernel(
    const float* __restrict__ qkf, float* __restrict__ kpart){
  int blk = blockIdx.x;                // 8*32
  int b = blk >> 5, ch = blk & 31;
  int c = threadIdx.x;
  const float* base = qkf + (((size_t)(b << 12)) + ch * 128) * 512 + 256 + c;
  float s = 0.f;
  for (int n = 0; n < 128; n++) s += base[(size_t)n * 512];
  kpart[((size_t)b * 32 + ch) * 256 + c] = s;
}

__global__ __launch_bounds__(256) void kmean_reduce_kernel(
    const float* __restrict__ kpart, float* __restrict__ km){
  int b = blockIdx.x;
  int c = threadIdx.x;
  float s = 0.f;
  for (int ch = 0; ch < 32; ch++) s += kpart[((size_t)b * 32 + ch) * 256 + c];
  km[b * 256 + c] = s * (1.f / 4096.f);
}

// ---------------------------------------------------------------------------
// RoPE on q,k (full C=256, 2D rotary, k_max=64) + z = 1/(q . kmean + 1e-6)
// ---------------------------------------------------------------------------
__global__ __launch_bounds__(256) void rope_z_kernel(
    const float* __restrict__ qkf, const float* __restrict__ kmean,
    __bf16* __restrict__ qrb, __bf16* __restrict__ krb, float* __restrict__ z)
{
  int m = blockIdx.x, c = threadIdx.x;
  int b = m >> 12, p = m & 4095;
  int y = p >> 6, x = p & 63;
  const float* row = qkf + (size_t)m * 512;
  float q  = row[c],        k  = row[256 + c];
  float qp = row[c ^ 1],    kp = row[256 + (c ^ 1)];
  int j = c >> 1;
  int jj = (j < 64) ? j : (j - 64);
  float pos = (j < 64) ? (float)y : (float)x;
  float theta = expf(-(float)jj * (9.2103403719761836f / 64.f)); // 10000^(-jj/64)
  float ang = pos * theta;
  float cs = cosf(ang), sn = sinf(ang);
  float qr, kr;
  if ((c & 1) == 0){ qr = q * cs - qp * sn; kr = k * cs - kp * sn; }
  else             { qr = qp * sn + q * cs; kr = kp * sn + k * cs; }
  qrb[(size_t)m * 256 + c] = (__bf16)qr;
  krb[(size_t)m * 256 + c] = (__bf16)kr;
  __shared__ float sh[256];
  sh[c] = q * kmean[b * 256 + c];
  __syncthreads();
  if (c < 8){
    float s = 0.f;
    for (int i = 0; i < 32; i++) s += sh[c * 32 + i];
    z[(size_t)m * 8 + c] = 1.f / (s + 1e-6f);
  }
}

// ---------------------------------------------------------------------------
// kv two-stage: partial kr^T @ v over 512-token chunks (LDS-staged outer
// product), then reduce + scale + transpose to bf16 (e-major, k contiguous).
// ---------------------------------------------------------------------------
__global__ __launch_bounds__(256) void kv_part_kernel(
    const __bf16* __restrict__ krb, const __bf16* __restrict__ vb,
    float* __restrict__ kvpart){
  int blk = blockIdx.x;                // 64 bh * 8 chunks
  int bh = blk >> 3, ch = blk & 7;
  int b = bh >> 3, h = bh & 7;
  int t = threadIdx.x;
  int e  = t & 31;
  int d0 = t >> 5;                     // 0..7
  float acc0 = 0.f, acc1 = 0.f, acc2 = 0.f, acc3 = 0.f;
  __shared__ float ks[8][32];
  __shared__ float vs[8][32];
  const size_t base = ((size_t)(b << 12)) * 256 + h * 32;
  int lr = t >> 5, lc = t & 31;
  int n0beg = ch * 512, n0end = n0beg + 512;
  for (int n0 = n0beg; n0 < n0end; n0 += 8){
    size_t idx = base + (size_t)(n0 + lr) * 256 + lc;
    ks[lr][lc] = (float)krb[idx];
    vs[lr][lc] = (float)vb[idx];
    __syncthreads();
    for (int i = 0; i < 8; i++){
      float vv = vs[i][e];
      acc0 += ks[i][d0     ] * vv;
      acc1 += ks[i][d0 +  8] * vv;
      acc2 += ks[i][d0 + 16] * vv;
      acc3 += ks[i][d0 + 24] * vv;
    }
    __syncthreads();
  }
  float* out = kvpart + ((size_t)ch * 64 + bh) * 1024;
  out[(d0     ) * 32 + e] = acc0;
  out[(d0 +  8) * 32 + e] = acc1;
  out[(d0 + 16) * 32 + e] = acc2;
  out[(d0 + 24) * 32 + e] = acc3;
}

__global__ __launch_bounds__(256) void kv_finish_kernel(
    const float* __restrict__ kvpart, __bf16* __restrict__ kvbT){
  int bh = blockIdx.x;                 // 0..63
  int t = threadIdx.x;
  int e  = t & 31;
  int d0 = t >> 5;
  const float sc = 1.f / 4096.f;
  for (int j = 0; j < 4; j++){
    int d = d0 + j * 8;
    float s = 0.f;
    for (int ch = 0; ch < 8; ch++)
      s += kvpart[((size_t)ch * 64 + bh) * 1024 + d * 32 + e];
    kvbT[((size_t)bh * 32 + e) * 32 + d] = (__bf16)(s * sc);
  }
}

// ---------------------------------------------------------------------------
// attn out: out[m, h*32+e] = z[m,h] * (qr[m, h*32:] @ kv[b,h])  (WMMA, K=32)
// Fused epilogue: + LePE (3x3 dw conv of v), * act_res, store bf16 a2.
// ---------------------------------------------------------------------------
__global__ __launch_bounds__(256) void attnout_kernel(
    const __bf16* __restrict__ qrb, const __bf16* __restrict__ kvbT,
    const float* __restrict__ z, const float* __restrict__ vbuf,
    const float* __restrict__ lepe_w, const float* __restrict__ lepe_b,
    const float* __restrict__ actres, __bf16* __restrict__ a2b)
{
  int blk = blockIdx.x;
  int bh = blk >> 4, mt = blk & 15;
  int b = bh >> 3, h = bh & 7;
  int tid = threadIdx.x, wid = tid >> 5, lane = tid & 31;
  int laneHalf = lane & 16;
  int rowBase = (b << 12) + mt * 256 + wid * 32;   // global token row
  int kb = laneHalf ? 16 : 0;
  v16bf b0 = *(const v16bf*)(kvbT + ((size_t)bh * 32 + (lane & 15)) * 32 + kb);
  v16bf b1 = *(const v16bf*)(kvbT + ((size_t)bh * 32 + 16 + (lane & 15)) * 32 + kb);
  v8f acc[2][2];
  int ka = laneHalf ? 16 : 0;
  for (int s = 0; s < 2; s++){
    int row = rowBase + s * 16 + (lane & 15);
    const __bf16* pa = qrb + (size_t)row * 256 + h * 32 + ka;
    v8bf lo = *(const v8bf*)pa;
    v8bf hi = *(const v8bf*)(pa + 8);
    v16bf av = SHUF16(lo, hi);
    v8f zf{};
    acc[s][0] = wmma_bf16(av, b0, zf);
    acc[s][1] = wmma_bf16(av, b1, zf);
  }
  int rAdd = laneHalf ? 8 : 0;
  for (int s = 0; s < 2; s++){
    for (int t = 0; t < 2; t++){
      int col = h * 32 + t * 16 + (lane & 15);
      float lw[9];
      for (int q2 = 0; q2 < 9; q2++) lw[q2] = lepe_w[col * 9 + q2];
      float lb = lepe_b[col];
      for (int i = 0; i < 8; i++){
        int row = rowBase + s * 16 + rAdd + i;
        float val = acc[s][t][i] * z[(size_t)row * 8 + h];
        int p = row & 4095, bb2 = row >> 12;
        int y = p >> 6, xx = p & 63;
        float lp = lb;
        for (int dy = -1; dy <= 1; dy++){
          int yy = y + dy; if (yy < 0 || yy >= 64) continue;
          for (int dx = -1; dx <= 1; dx++){
            int xx2 = xx + dx; if (xx2 < 0 || xx2 >= 64) continue;
            lp += vbuf[(((size_t)(bb2 << 12)) + yy * 64 + xx2) * 256 + col] *
                  lw[(dy + 1) * 3 + (dx + 1)];
          }
        }
        size_t oi = (size_t)row * 256 + col;
        a2b[oi] = (__bf16)((val + lp) * actres[oi]);
      }
    }
  }
}

// ---------------------------------------------------------------------------
// Generic bf16 WMMA GEMM: out = epilogue(A[M,K] @ WT[N,K]^T + bias)
// Block: 256 threads = 8 waves (4M x 2N), tile 128x64, wave tile 32x32.
// Weight tile (64x32 bf16) staged into LDS by TDM each K-step; B fragments
// come from LDS, A fragments from global (token activations).
// MODE: 0 f32 store, 1 SiLU f32, 2 elu+1 f32, 3 GELU bf16,
//       4 residual add f32, 5 residual add + scatter to NCHW f32
// ---------------------------------------------------------------------------
template<int MODE>
__global__ __launch_bounds__(256) void gemm_kernel(
    const __bf16* __restrict__ A, const __bf16* __restrict__ WT,
    const float* __restrict__ bias, int N, int K,
    const float* __restrict__ res, float* __restrict__ outF,
    __bf16* __restrict__ outB)
{
  __shared__ alignas(32) __bf16 tileB[64 * 32];     // 4 KB weight tile
  int nbn = N >> 6;
  int bm = blockIdx.x / nbn;
  int bn = blockIdx.x - bm * nbn;
  int tid = threadIdx.x, wid = tid >> 5, lane = tid & 31;
  int wm = wid >> 1, wn = wid & 1;
  int rowBase = bm * 128 + wm * 32;
  int colBase = bn * 64 + wn * 32;
  int laneHalf = lane & 16;
  int r0 = rowBase + (lane & 15);
  int lc0 = wn * 32 + (lane & 15);                  // local col in 64-wide tile
  unsigned int ldsOff = (unsigned int)(uintptr_t)&tileB[0]; // low32 = LDS offset
  v8f acc00{}, acc01{}, acc10{}, acc11{};
  for (int kk = 0; kk < K; kk += 32){
    __syncthreads();                                // previous tile fully consumed
    if (wid == 0){
      tdm_load_tile_b(ldsOff, WT + (size_t)(bn * 64) * K + kk, K, N);
      __builtin_amdgcn_s_wait_tensorcnt(0);
    }
    __syncthreads();                                // tile visible to all waves
    // A fragments (global, bf16 activations)
    int ka = kk + (laneHalf ? 16 : 0);
    const __bf16* pa0 = A + (size_t)r0 * K + ka;
    v8bf lo0 = *(const v8bf*)pa0;
    v8bf hi0 = *(const v8bf*)(pa0 + 8);
    v16bf a0 = SHUF16(lo0, hi0);
    const __bf16* pa1 = pa0 + (size_t)16 * K;
    v8bf lo1 = *(const v8bf*)pa1;
    v8bf hi1 = *(const v8bf*)(pa1 + 8);
    v16bf a1 = SHUF16(lo1, hi1);
    // B fragments from LDS
    int kw = laneHalf ? 16 : 0;
    v16bf b0 = *(const v16bf*)&tileB[lc0 * 32 + kw];
    v16bf b1 = *(const v16bf*)&tileB[(lc0 + 16) * 32 + kw];
    acc00 = wmma_bf16(a0, b0, acc00);
    acc01 = wmma_bf16(a0, b1, acc01);
    acc10 = wmma_bf16(a1, b0, acc10);
    acc11 = wmma_bf16(a1, b1, acc11);
  }
  int rAdd = laneHalf ? 8 : 0;
  for (int s = 0; s < 2; s++){
    for (int t = 0; t < 2; t++){
      v8f acc = (s == 0) ? (t == 0 ? acc00 : acc01) : (t == 0 ? acc10 : acc11);
      int col = colBase + t * 16 + (lane & 15);
      float bv = bias[col];
      for (int i = 0; i < 8; i++){
        int row = rowBase + s * 16 + rAdd + i;
        float v = acc[i] + bv;
        size_t oi = (size_t)row * N + col;
        if (MODE == 0){
          outF[oi] = v;
        } else if (MODE == 1){
          outF[oi] = v / (1.f + expf(-v));                         // SiLU
        } else if (MODE == 2){
          outF[oi] = (v > 0.f) ? (v + 1.f) : expf(v);              // elu + 1
        } else if (MODE == 3){
          outB[oi] = (__bf16)(0.5f * v * (1.f + erff(v * 0.70710678118f))); // GELU
        } else if (MODE == 4){
          outF[oi] = res[oi] + v;                                  // + shortcut
        } else {                                                   // 5: + residual, NCHW scatter
          float r = res[oi] + v;
          int bb2 = row >> 12, p = row & 4095;
          outF[(((size_t)bb2 * 256 + col) << 12) + p] = r;
        }
      }
    }
  }
}

// ---------------------------------------------------------------------------
// Host orchestration
// ---------------------------------------------------------------------------
extern "C" void kernel_launch(void* const* d_in, const int* in_sizes, int n_in,
                              void* d_out, int out_size, void* d_ws, size_t ws_size,
                              hipStream_t stream)
{
  (void)in_sizes; (void)n_in; (void)out_size; (void)ws_size;
  const float* x         = (const float*)d_in[0];
  const float* cpe1_w    = (const float*)d_in[1];
  const float* cpe1_b    = (const float*)d_in[2];
  const float* n1_g      = (const float*)d_in[3];
  const float* n1_b      = (const float*)d_in[4];
  const float* inproj_w  = (const float*)d_in[5];
  const float* inproj_b  = (const float*)d_in[6];
  const float* actproj_w = (const float*)d_in[7];
  const float* actproj_b = (const float*)d_in[8];
  const float* dwc_w     = (const float*)d_in[9];
  const float* dwc_b     = (const float*)d_in[10];
  const float* qk_w      = (const float*)d_in[11];
  const float* qk_b      = (const float*)d_in[12];
  const float* lepe_w    = (const float*)d_in[13];
  const float* lepe_b    = (const float*)d_in[14];
  const float* outproj_w = (const float*)d_in[15];
  const float* outproj_b = (const float*)d_in[16];
  const float* cpe2_w    = (const float*)d_in[17];
  const float* cpe2_b    = (const float*)d_in[18];
  const float* n2_g      = (const float*)d_in[19];
  const float* n2_b      = (const float*)d_in[20];
  const float* fc1_w     = (const float*)d_in[21];
  const float* fc1_b     = (const float*)d_in[22];
  const float* fc2_w     = (const float*)d_in[23];
  const float* fc2_b     = (const float*)d_in[24];

  const int M = 32768;                // B * H * W tokens
  const size_t MC4 = (size_t)M * 256 * 4;
  const size_t MC2 = (size_t)M * 256 * 2;

  char* wsb = (char*)d_ws;
  size_t off = 0;
  auto alloc = [&](size_t bytes) -> size_t {
    size_t o = off; off = (off + bytes + 255) & ~(size_t)255; return o;
  };
  size_t o_t      = alloc(MC4);                       // t (residual), f32
  size_t o_tn     = alloc(MC2);                       // LN out bf16 (reused as tn2)
  size_t o_actres = alloc(MC4);                       // silu(actproj) f32
  size_t o_ti     = alloc(MC4);                       // inproj out f32 (reused as t2)
  size_t o_v      = alloc(MC4);                       // v f32 (post dwc+silu)
  size_t o_vb     = alloc(MC2);                       // v bf16
  size_t o_qkf    = alloc((size_t)M * 512 * 4);       // [q|k] f32 (elu+1)
  size_t o_qrb    = alloc(MC2);                       // roped q bf16
  size_t o_krb    = alloc(MC2);                       // roped k bf16
  size_t o_z      = alloc((size_t)M * 8 * 4);         // z f32
  size_t o_kmean  = alloc((size_t)8 * 256 * 4);       // kmean f32
  size_t o_kpart  = alloc((size_t)8 * 32 * 256 * 4);  // kmean partials
  size_t o_kvb    = alloc((size_t)64 * 32 * 32 * 2);  // kv bf16 (e-major)
  size_t o_kvpart = alloc((size_t)8 * 64 * 1024 * 4); // kv partials f32
  size_t o_a2b    = alloc(MC2);                       // (ta+lepe)*actres bf16
  size_t o_h1     = alloc((size_t)M * 1024 * 2);      // gelu(fc1) bf16
  size_t o_wact   = alloc((size_t)256 * 256 * 2);
  size_t o_win    = alloc((size_t)256 * 256 * 2);
  size_t o_wqk    = alloc((size_t)256 * 512 * 2);
  size_t o_wout   = alloc((size_t)256 * 256 * 2);
  size_t o_wfc1   = alloc((size_t)256 * 1024 * 2);
  size_t o_wfc2   = alloc((size_t)1024 * 256 * 2);

  float*  t      = (float*)(wsb + o_t);
  __bf16* tn     = (__bf16*)(wsb + o_tn);
  float*  actres = (float*)(wsb + o_actres);
  float*  ti     = (float*)(wsb + o_ti);    // later reused as t2
  float*  v      = (float*)(wsb + o_v);
  __bf16* vb     = (__bf16*)(wsb + o_vb);
  float*  qkf    = (float*)(wsb + o_qkf);
  __bf16* qrb    = (__bf16*)(wsb + o_qrb);
  __bf16* krb    = (__bf16*)(wsb + o_krb);
  float*  z      = (float*)(wsb + o_z);
  float*  kmean  = (float*)(wsb + o_kmean);
  float*  kpart  = (float*)(wsb + o_kpart);
  __bf16* kvbT   = (__bf16*)(wsb + o_kvb);
  float*  kvpart = (float*)(wsb + o_kvpart);
  __bf16* a2b    = (__bf16*)(wsb + o_a2b);
  __bf16* h1     = (__bf16*)(wsb + o_h1);
  __bf16* wact   = (__bf16*)(wsb + o_wact);
  __bf16* win    = (__bf16*)(wsb + o_win);
  __bf16* wqk    = (__bf16*)(wsb + o_wqk);
  __bf16* wout   = (__bf16*)(wsb + o_wout);
  __bf16* wfc1   = (__bf16*)(wsb + o_wfc1);
  __bf16* wfc2   = (__bf16*)(wsb + o_wfc2);

  // 1) weights -> transposed bf16
  wt_kernel<<<256,  256, 0, stream>>>(actproj_w, wact, 256, 256);
  wt_kernel<<<256,  256, 0, stream>>>(inproj_w,  win,  256, 256);
  wt_kernel<<<512,  256, 0, stream>>>(qk_w,      wqk,  256, 512);
  wt_kernel<<<256,  256, 0, stream>>>(outproj_w, wout, 256, 256);
  wt_kernel<<<1024, 256, 0, stream>>>(fc1_w,     wfc1, 256, 1024);
  wt_kernel<<<1024, 256, 0, stream>>>(fc2_w,     wfc2, 1024, 256);

  // 2) CPE1 + LN1
  cpe_ln_kernel<<<M, 256, 0, stream>>>(x, 1, cpe1_w, cpe1_b, n1_g, n1_b, t, tn);

  // 3) actproj (SiLU), 4) inproj
  gemm_kernel<1><<<256 * 4, 256, 0, stream>>>(tn, wact, actproj_b, 256, 256, nullptr, actres, nullptr);
  gemm_kernel<0><<<256 * 4, 256, 0, stream>>>(tn, win,  inproj_b,  256, 256, nullptr, ti,     nullptr);

  // 5) depthwise conv + SiLU -> v
  dwc_silu_kernel<<<M, 256, 0, stream>>>(ti, dwc_w, dwc_b, v, vb);

  // 6) qk projection (elu+1)
  gemm_kernel<2><<<256 * 8, 256, 0, stream>>>(vb, wqk, qk_b, 512, 256, nullptr, qkf, nullptr);

  // 7) kmean (two-stage), 8) RoPE + z, 9) kv (two-stage)
  kmean_part_kernel<<<256, 256, 0, stream>>>(qkf, kpart);
  kmean_reduce_kernel<<<8, 256, 0, stream>>>(kpart, kmean);
  rope_z_kernel<<<M, 256, 0, stream>>>(qkf, kmean, qrb, krb, z);
  kv_part_kernel<<<512, 256, 0, stream>>>(krb, vb, kvpart);
  kv_finish_kernel<<<64, 256, 0, stream>>>(kvpart, kvbT);

  // 10) attention out (WMMA) + LePE + act gating
  attnout_kernel<<<1024, 256, 0, stream>>>(qrb, kvbT, z, v, lepe_w, lepe_b, actres, a2b);

  // 11) outproj + shortcut (in-place into t)
  gemm_kernel<4><<<256 * 4, 256, 0, stream>>>(a2b, wout, outproj_b, 256, 256, t, t, nullptr);

  // 12) CPE2 + LN2 (t -> t2 in o_ti, tn2 in o_tn)
  cpe_ln_kernel<<<M, 256, 0, stream>>>(t, 0, cpe2_w, cpe2_b, n2_g, n2_b, ti, tn);

  // 13) fc1 + GELU, 14) fc2 + residual, scatter to NCHW output
  gemm_kernel<3><<<256 * 16, 256, 0, stream>>>(tn, wfc1, fc1_b, 1024, 256, nullptr, nullptr, h1);
  gemm_kernel<5><<<256 * 4, 256, 0, stream>>>(h1, wfc2, fc2_b, 256, 1024, ti, (float*)d_out, nullptr);
}
